// MultiHeadAttention_29566554865733
// MI455X (gfx1250) — compile-verified, matrix operand order checked by reading, not run on hardware
//
#include <hip/hip_runtime.h>
#include <hip/hip_bf16.h>

// ---------------------------------------------------------------------------
// MHA forward on gfx1250 (CDNA5): bf16 WMMA + TDM (tensor_load_to_lds).
//   B=2, S=2048, DIM=2048, H=16, HD=128
// ---------------------------------------------------------------------------

#define BATCH 2
#define SEQ   2048
#define DIM   2048
#define HEADS 16
#define HD    128

typedef __bf16 bf16_t;
typedef bf16_t v16bf __attribute__((ext_vector_type(16)));
typedef bf16_t v8bf  __attribute__((ext_vector_type(8)));
typedef float  v8f   __attribute__((ext_vector_type(8)));

typedef unsigned int u32x4 __attribute__((ext_vector_type(4)));
typedef int          i32x8 __attribute__((ext_vector_type(8)));
typedef int          i32x4 __attribute__((ext_vector_type(4)));

union FragBF {
    v16bf v;
    v8bf  h[2];
    unsigned short us[16];
};

__device__ __forceinline__ unsigned short f2bf(float x) {
    unsigned int u = __float_as_uint(x);
    u += 0x7FFFu + ((u >> 16) & 1u);       // round-to-nearest-even
    return (unsigned short)(u >> 16);
}
__device__ __forceinline__ float bf2f(unsigned short b) {
    return __uint_as_float(((unsigned int)b) << 16);
}

// Load one 16x32 bf16 A/B fragment (ISA 7.12.2 layout for 16-bit 16x32):
//   lane L: row = L&15, half = L>>4
//   VGPR 0..3 hold K = half*8 + {0..7}; VGPR 4..7 hold K = 16 + half*8 + {0..7}
// Two contiguous 16-byte loads from row-major memory (global OR LDS).
__device__ __forceinline__ v16bf load_frag(const unsigned short* base,
                                           int row0, int ld, int k0, int lane) {
    int half = (lane >> 4) & 1;
    int r    = lane & 15;
    const unsigned short* p = base + (size_t)(row0 + r) * (size_t)ld + k0 + half * 8;
    FragBF f;
    f.h[0] = *reinterpret_cast<const v8bf*>(p);       // K = half*8 + 0..7
    f.h[1] = *reinterpret_cast<const v8bf*>(p + 16);  // K = 16 + half*8 + 0..7
    return f.v;
}

// ---------------------------------------------------------------------------
// TDM: 2D tile load global->LDS.  Descriptor per ISA 8.3/8.4:
//  group0: count=1 | lds_addr | global_addr(57b) | type=2
//  group1: data_size=1(2B), tensor_dim0=tile_d0, tensor_dim1=tile_d1,
//          tile_dim0/1, tensor_dim0_stride (data_size units)
// This toolchain exposes the 6-arg builtin:
//  (u32x4 g0, i32x8 g1, i32x4 g2, i32x4 g3, i32x8 extra, i32 cpol)
// ---------------------------------------------------------------------------
__device__ __forceinline__ void tdm_load_2d(const unsigned short* gptr,
                                            unsigned lds_off,
                                            unsigned tile_d0, unsigned tile_d1,
                                            unsigned long long stride0) {
    unsigned long long ga = (unsigned long long)(uintptr_t)gptr;
    u32x4 g0;
    g0[0] = 1u;                                            // count=1 (valid), user mode
    g0[1] = lds_off;                                       // LDS byte address
    g0[2] = (unsigned)(ga & 0xffffffffu);                  // global_addr[31:0]
    g0[3] = (unsigned)((ga >> 32) & 0x01ffffffu) | (2u << 30);  // addr[56:32] | type=2

    i32x8 g1;
    g1[0] = (int)(1u << 16);                               // data_size=1 (2 bytes)
    g1[1] = (int)((tile_d0 & 0xffffu) << 16);              // tensor_dim0[15:0] @ [63:48]
    g1[2] = (int)((tile_d0 >> 16) | ((tile_d1 & 0xffffu) << 16)); // dim0[31:16] | dim1[15:0]
    g1[3] = (int)((tile_d1 >> 16) | ((tile_d0 & 0xffffu) << 16)); // dim1[31:16] | tile_dim0
    g1[4] = (int)(tile_d1 & 0xffffu);                      // tile_dim1 (tile_dim2=0)
    g1[5] = (int)(unsigned)(stride0 & 0xffffffffu);        // tensor_dim0_stride[31:0]
    g1[6] = (int)(unsigned)((stride0 >> 32) & 0xffffu);    // stride0[47:32] (stride1=0)
    g1[7] = 0;

    i32x4 gz  = {0, 0, 0, 0};                              // groups 2/3 unused (2D)
    i32x8 gz8 = {0, 0, 0, 0, 0, 0, 0, 0};
    __builtin_amdgcn_tensor_load_to_lds(g0, g1, gz, gz, gz8, 0);
}

// ---------------------------------------------------------------------------
// fp32 -> bf16 conversion (grid-stride)
// ---------------------------------------------------------------------------
__global__ __launch_bounds__(256) void cvt_f32_bf16(const float* __restrict__ src,
                                                    unsigned short* __restrict__ dst,
                                                    int n) {
    int i = blockIdx.x * blockDim.x + threadIdx.x;
    int stride = gridDim.x * blockDim.x;
    for (; i < n; i += stride) dst[i] = f2bf(src[i]);
}

// ---------------------------------------------------------------------------
// C[M,N] = A[M,K] * B[N,K]^T   (both bf16, K-contiguous for both operands)
// one wave -> 32x64 output tile: 6 fragment loads per 8 WMMAs
// ---------------------------------------------------------------------------
template <bool F32OUT>
__global__ __launch_bounds__(256) void gemm_bf16(const unsigned short* __restrict__ A,
                                                 const unsigned short* __restrict__ B,
                                                 unsigned short* __restrict__ Cb,
                                                 float* __restrict__ Cf,
                                                 int M, int N, int K) {
    int lane = threadIdx.x & 31;
    int wave = blockIdx.x * (blockDim.x >> 5) + (threadIdx.x >> 5);
    int ntn  = N >> 6;                 // N / 64
    int mt   = wave / ntn;
    int nt   = wave - mt * ntn;
    if (mt * 32 >= M) return;
    int m0 = mt * 32, n0 = nt * 64;

    v8f acc[2][4] = {};
    for (int k0 = 0; k0 < K; k0 += 32) {
        v16bf a0 = load_frag(A, m0,      K, k0, lane);
        v16bf a1 = load_frag(A, m0 + 16, K, k0, lane);
        if (k0 + 32 < K) {
            // CDNA5 global_prefetch_b8 for the next A strips
            __builtin_prefetch(A + (size_t)(m0 + (lane & 15)) * (size_t)K + k0 + 32, 0, 0);
            __builtin_prefetch(A + (size_t)(m0 + 16 + (lane & 15)) * (size_t)K + k0 + 32, 0, 0);
        }
#pragma unroll
        for (int j = 0; j < 4; ++j) {
            v16bf bfrag = load_frag(B, n0 + 16 * j, K, k0, lane);
            acc[0][j] = __builtin_amdgcn_wmma_f32_16x16x32_bf16(
                false, a0, false, bfrag, (short)0, acc[0][j], false, false);
            acc[1][j] = __builtin_amdgcn_wmma_f32_16x16x32_bf16(
                false, a1, false, bfrag, (short)0, acc[1][j], false, false);
        }
    }

    int half = lane >> 4, col = lane & 15;
#pragma unroll
    for (int mi = 0; mi < 2; ++mi) {
#pragma unroll
        for (int j = 0; j < 4; ++j) {
#pragma unroll
            for (int i = 0; i < 8; ++i) {
                size_t off = (size_t)(m0 + 16 * mi + i + 8 * half) * (size_t)N
                           + n0 + 16 * j + col;
                if (F32OUT) Cf[off] = acc[mi][j][i];
                else        Cb[off] = f2bf(acc[mi][j][i]);
            }
        }
    }
}

// ---------------------------------------------------------------------------
// Non-interleaved (chunk-half) RoPE, in place on bf16 [B,S,H,HD]
// ---------------------------------------------------------------------------
__global__ __launch_bounds__(256) void rope_kernel(unsigned short* __restrict__ X) {
    int idx = blockIdx.x * blockDim.x + threadIdx.x;
    const int HALF = HD / 2;
    int total = BATCH * SEQ * HEADS * HALF;
    if (idx >= total) return;
    int i = idx % HALF;
    int t = idx / HALF;
    int h = t % HEADS; t /= HEADS;
    int s = t % SEQ;
    int b = t / SEQ;
    size_t base = (((size_t)b * SEQ + s) * HEADS + h) * HD;
    float freq = (float)s * __expf(-(float)i * (2.0f / (float)HD) * 9.210340371976184f);
    float c = __cosf(freq), sn = __sinf(freq);
    float x1 = bf2f(X[base + i]);
    float x2 = bf2f(X[base + i + HALF]);
    X[base + i]        = f2bf(x1 * c - x2 * sn);
    X[base + i + HALF] = f2bf(x1 * sn + x2 * c);
}

// ---------------------------------------------------------------------------
// V [B,S,H,HD] -> VT [B,H,HD,S]  (so P*V B-fragments are K-contiguous)
// ---------------------------------------------------------------------------
__global__ __launch_bounds__(256) void transpose_v(const unsigned short* __restrict__ V,
                                                   unsigned short* __restrict__ VT) {
    int i = blockIdx.x * blockDim.x + threadIdx.x;
    int total = BATCH * SEQ * HEADS * HD;
    if (i >= total) return;
    int d = i % HD; int t = i / HD;
    int h = t % HEADS; t /= HEADS;
    int s = t % SEQ;
    int b = t / SEQ;
    VT[(((size_t)b * HEADS + h) * HD + d) * SEQ + s] = V[i];
}

// ---------------------------------------------------------------------------
// Causal flash attention: ONE wave per workgroup, one (b,h,16-row q tile) each.
// K/V tiles are double-buffered in LDS via TDM (tensor_load_to_lds) and
// synchronized with s_wait_tensorcnt; fragments then come from LDS.
// Per chunk of 32 keys: 8 WMMA (Q*K^T) + online softmax + 8 WMMA (P*V^T).
// ---------------------------------------------------------------------------
#define KOFF(buf) ((unsigned)(buf) * 8192u)            // 32x128 bf16 = 8KB
#define VOFF(buf) (16384u + (unsigned)(buf) * 8192u)   // 128x32 bf16 = 8KB
#define POFF      32768u                               // 16x33 f32 staging

__global__ __launch_bounds__(32) void attn_kernel(const unsigned short* __restrict__ Q,
                                                  const unsigned short* __restrict__ Kb,
                                                  const unsigned short* __restrict__ VT,
                                                  unsigned short* __restrict__ O) {
    __shared__ __align__(16) unsigned char smem[34944];

    int lane = threadIdx.x & 31;
    int wave = blockIdx.x;
    const int qtiles = SEQ / 16;        // 128
    if (wave >= BATCH * HEADS * qtiles) return;
    int qt = wave % qtiles;
    int t  = wave / qtiles;
    int h  = t % HEADS;
    int b  = t / HEADS;

    int half = lane >> 4;
    int col  = lane & 15;
    int q0   = qt * 16;

    const unsigned short* Qp = Q  + ((size_t)b * SEQ) * DIM + h * HD;
    const unsigned short* Kp = Kb + ((size_t)b * SEQ) * DIM + h * HD;
    const unsigned short* Vp = VT + ((size_t)(b * HEADS + h) * HD) * SEQ;
    float* Pst = (float*)(smem + POFF);                 // [16][33]

    // Q fragments: head dim 128 = 4 chunks of K=32 (row stride DIM)
    v16bf qf[4];
#pragma unroll
    for (int d = 0; d < 4; ++d) qf[d] = load_frag(Qp, q0, DIM, d * 32, lane);

    v8f oacc[8] = {};
    float mrow[8], lrow[8];
#pragma unroll
    for (int i = 0; i < 8; ++i) { mrow[i] = -1e30f; lrow[i] = 0.0f; }

    int nchunks = (q0 + 16 + 31) >> 5;

    // prologue: TDM-stage chunk 0 into buffer 0 (2 tensor loads)
    tdm_load_2d(Kp,          KOFF(0), HD, 32, (unsigned long long)DIM);
    tdm_load_2d(Vp,          VOFF(0), 32, HD, (unsigned long long)SEQ);

    for (int c = 0; c < nchunks; ++c) {
        int kb0 = c * 32;
        int buf = c & 1;
        if (c + 1 < nchunks) {
            // stream next chunk into the other buffer while computing this one
            int kn = (c + 1) * 32;
            tdm_load_2d(Kp + (size_t)kn * DIM, KOFF(buf ^ 1), HD, 32,
                        (unsigned long long)DIM);
            tdm_load_2d(Vp + kn,               VOFF(buf ^ 1), 32, HD,
                        (unsigned long long)SEQ);
            __builtin_amdgcn_s_wait_tensorcnt(2);  // current buffer resident
        } else {
            __builtin_amdgcn_s_wait_tensorcnt(0);
        }
        const unsigned short* Ks = (const unsigned short*)(smem + KOFF(buf)); // [32][128]
        const unsigned short* Vs = (const unsigned short*)(smem + VOFF(buf)); // [128][32]

        v8f sc0 = {}, sc1 = {};
#pragma unroll
        for (int d = 0; d < 4; ++d) {
            v16bf b0 = load_frag(Ks, 0,  HD, d * 32, lane);
            v16bf b1 = load_frag(Ks, 16, HD, d * 32, lane);
            sc0 = __builtin_amdgcn_wmma_f32_16x16x32_bf16(false, qf[d], false, b0,
                                                          (short)0, sc0, false, false);
            sc1 = __builtin_amdgcn_wmma_f32_16x16x32_bf16(false, qf[d], false, b1,
                                                          (short)0, sc1, false, false);
        }
        const float scale = 0.0883883476483184f;   // 1/sqrt(128)
        float pr0[8], pr1[8];
#pragma unroll
        for (int i = 0; i < 8; ++i) {
            int q = q0 + i + 8 * half;
            float s0 = sc0[i] * scale;
            float s1 = sc1[i] * scale;
            if (kb0 + col > q)      s0 = -1e30f;    // causal mask
            if (kb0 + 16 + col > q) s1 = -1e30f;
            float mt = fmaxf(s0, s1);
#pragma unroll
            for (int m = 1; m < 16; m <<= 1) mt = fmaxf(mt, __shfl_xor(mt, m, 32));
            float mnew  = fmaxf(mrow[i], mt);
            float alpha = __expf(mrow[i] - mnew);
            float p0 = __expf(s0 - mnew);
            float p1 = __expf(s1 - mnew);
            float rs = p0 + p1;
#pragma unroll
            for (int m = 1; m < 16; m <<= 1) rs += __shfl_xor(rs, m, 32);
            lrow[i] = lrow[i] * alpha + rs;
            mrow[i] = mnew;
            pr0[i] = p0; pr1[i] = p1;
#pragma unroll
            for (int n = 0; n < 8; ++n) oacc[n][i] *= alpha;
        }
        // stage P (C-layout) -> LDS -> reload in A-fragment layout
#pragma unroll
        for (int i = 0; i < 8; ++i) {
            Pst[(i + 8 * half) * 33 + col]      = pr0[i];
            Pst[(i + 8 * half) * 33 + 16 + col] = pr1[i];
        }
        FragBF pf;
#pragma unroll
        for (int e = 0; e < 16; ++e) {
            int kk = ((e >= 8) ? 16 : 0) + half * 8 + (e & 7);
            pf.us[e] = f2bf(Pst[col * 33 + kk]);
        }
        // out += P (16x32) * V^T tile (rows = head-dim cols, K = keys)
#pragma unroll
        for (int n = 0; n < 8; ++n) {
            v16bf vf = load_frag(Vs, n * 16, 32, 0, lane);
            oacc[n] = __builtin_amdgcn_wmma_f32_16x16x32_bf16(false, pf.v, false, vf,
                                                              (short)0, oacc[n], false, false);
        }
    }

    // normalize and scatter to [B,S,H,HD] bf16
#pragma unroll
    for (int i = 0; i < 8; ++i) {
        float inv = 1.0f / lrow[i];
        int q = q0 + i + 8 * half;
        unsigned short* orow = O + ((size_t)(b * SEQ + q)) * DIM + h * HD;
#pragma unroll
        for (int n = 0; n < 8; ++n) orow[n * 16 + col] = f2bf(oacc[n][i] * inv);
    }
}

// ---------------------------------------------------------------------------
// Host-side launcher
// ---------------------------------------------------------------------------
extern "C" void kernel_launch(void* const* d_in, const int* in_sizes, int n_in,
                              void* d_out, int out_size, void* d_ws, size_t ws_size,
                              hipStream_t stream) {
    const float* x  = (const float*)d_in[0];
    const float* Wq = (const float*)d_in[1];
    const float* Wk = (const float*)d_in[2];
    const float* Wv = (const float*)d_in[3];
    const float* Wo = (const float*)d_in[4];
    float* out = (float*)d_out;
    unsigned short* ws = (unsigned short*)d_ws;

    const size_t NX = (size_t)BATCH * SEQ * DIM;  // 8,388,608
    const size_t NW = (size_t)DIM * DIM;          // 4,194,304

    unsigned short* XB  = ws;
    unsigned short* WQb = XB  + NX;
    unsigned short* WKb = WQb + NW;
    unsigned short* WVb = WKb + NW;
    unsigned short* WOb = WVb + NW;
    unsigned short* Qb  = WOb + NW;
    unsigned short* Kbf = Qb  + NX;
    unsigned short* Vb  = Kbf + NX;
    unsigned short* VTb = Vb  + NX;
    unsigned short* ATb = VTb + NX;   // total: 128 MB of bf16 staging

    // 1) fp32 -> bf16
    cvt_f32_bf16<<<1024, 256, 0, stream>>>(x,  XB,  (int)NX);
    cvt_f32_bf16<<<1024, 256, 0, stream>>>(Wq, WQb, (int)NW);
    cvt_f32_bf16<<<1024, 256, 0, stream>>>(Wk, WKb, (int)NW);
    cvt_f32_bf16<<<1024, 256, 0, stream>>>(Wv, WVb, (int)NW);
    cvt_f32_bf16<<<1024, 256, 0, stream>>>(Wo, WOb, (int)NW);

    // 2) projections: [4096,2048] x [2048,2048]^T, one wave per 32x64 tile
    const int M = BATCH * SEQ;                         // 4096
    const int gemmWaves  = (M / 32) * (DIM / 64);      // 4096
    const int gemmBlocks = gemmWaves / 8;              // 512
    gemm_bf16<false><<<gemmBlocks, 256, 0, stream>>>(XB, WQb, Qb,  nullptr, M, DIM, DIM);
    gemm_bf16<false><<<gemmBlocks, 256, 0, stream>>>(XB, WKb, Kbf, nullptr, M, DIM, DIM);
    gemm_bf16<false><<<gemmBlocks, 256, 0, stream>>>(XB, WVb, Vb,  nullptr, M, DIM, DIM);

    // 3) RoPE in place on Q, K
    const int ropeTotal  = BATCH * SEQ * HEADS * (HD / 2);  // 4,194,304
    rope_kernel<<<ropeTotal / 256, 256, 0, stream>>>(Qb);
    rope_kernel<<<ropeTotal / 256, 256, 0, stream>>>(Kbf);

    // 4) V transpose for K-contiguous P*V fragments
    transpose_v<<<(int)(NX / 256), 256, 0, stream>>>(Vb, VTb);

    // 5) causal flash attention: 1 wave per workgroup (TDM double buffering)
    const int attnWaves = BATCH * HEADS * (SEQ / 16);  // 4096
    attn_kernel<<<attnWaves, 32, 0, stream>>>(Qb, Kbf, VTb, ATb);

    // 6) output projection, fp32 result straight to d_out
    gemm_bf16<true><<<gemmBlocks, 256, 0, stream>>>(ATb, WOb, nullptr, out, M, DIM, DIM);
}
